// AttentionAggregator_66915590472577
// MI455X (gfx1250) — matrix operand order repaired and checked by reading, not compiled, not run on hardware
//
#include <hip/hip_runtime.h>
#include <hip/hip_fp16.h>

// ---------------- problem constants ----------------
#define B_TOT   32768
#define KNBR    32
#define DIM     128
#define FDIM    128
#define HEADS   4
#define KP1     33                   // K+1 (neighbors + self)
#define TILE_B  16
#define ROWS    (TILE_B * KNBR)      // 512 neighbor rows per tile
#define NTHR    512                  // 16 wave32s
#define ROWP    132                  // padded LDS row stride (floats): 16B-aligned rows
#define KTOT    (HEADS * DIM)        // 512 = K dim of final GEMM
#define YSTR    516                  // padded yh row stride (halfs): 8B-aligned rows

typedef _Float16 v16h  __attribute__((ext_vector_type(16)));
typedef float    v8f   __attribute__((ext_vector_type(8)));
typedef float    f32x4 __attribute__((ext_vector_type(4)));
typedef _Float16 f16x4 __attribute__((ext_vector_type(4)));

// ---------------- LDS layout (float offsets) ----------------
#define NB_F     0
#define SELF_F   (NB_F + ROWS*ROWP)              // 67584
#define VS_F     (SELF_F + TILE_B*ROWP)          // 69696
#define VN_F     (VS_F + KTOT)                   // 70208
#define NDOT_F   (VN_F + KTOT)                   // 70720
#define SCS_F    (NDOT_F + ROWS*HEADS)           // 72768
#define SNS_F    (SCS_F + TILE_B*HEADS)          // 72832
#define ALPHA_F  (SNS_F + TILE_B*HEADS)          // 72896
#define YH_F     (ALPHA_F + TILE_B*HEADS*KP1)    // 75008 (x4 bytes: 16B aligned)
#define OUT_F    (YH_F + (TILE_B*YSTR)/2)        // 79136
#define RN_F     (OUT_F + TILE_B*FDIM)           // 81184
#define SMEM_F   (RN_F + TILE_B)                 // 81200
#define SMEM_BYTES (SMEM_F * 4)                  // 324800 B <= 320 KB/WGP

// ---------------- TDM (Tensor Data Mover) ----------------
#if defined(__gfx1250__) && __has_builtin(__builtin_amdgcn_tensor_load_to_lds) && __has_builtin(__builtin_amdgcn_s_wait_tensorcnt)
#define USE_TDM 1
#endif

typedef unsigned int uvec4 __attribute__((ext_vector_type(4)));
typedef int          ivec4 __attribute__((ext_vector_type(4)));
typedef int          ivec8 __attribute__((ext_vector_type(8)));

#ifdef USE_TDM
// 2D fp32 tile load: rows x 128 floats (row stride 128 in memory),
// LDS rows padded to 132 floats via D# pad (interval=128 dw, amount=4 dw).
__device__ __forceinline__ void tdm_load_tile_f32(unsigned lds_byte,
                                                  const float* gsrc,
                                                  unsigned rows) {
  unsigned long long ga = (unsigned long long)(uintptr_t)gsrc;
  uvec4 g0;
  g0.x = 1u;                                       // count=1 (valid), user mode
  g0.y = lds_byte;                                 // lds_addr (bytes)
  g0.z = (unsigned)ga;                             // global_addr[31:0]
  g0.w = ((unsigned)(ga >> 32) & 0x01FFFFFFu) | (2u << 30); // addr[56:32] | type=2
  ivec8 g1;
  g1[0] = (int)((2u << 16) | (1u << 20) | (6u << 22) | (3u << 25));
  //            data_size=4B | pad_en | interval=128dw | pad_amount=4dw
  g1[1] = (int)(128u << 16);                       // tensor_dim0 = 128 (low16)
  g1[2] = (int)((rows & 0xFFFFu) << 16);           // tensor_dim1 (low16)
  g1[3] = (int)(128u << 16);                       // tile_dim0 = 128
  g1[4] = (int)(rows & 0xFFFFu);                   // tile_dim1 = rows
  g1[5] = 128;                                     // tensor_dim0_stride = 128 elems
  g1[6] = 0;
  g1[7] = 0;
  ivec4 z4 = {0, 0, 0, 0};
#if __clang_major__ >= 23
  ivec8 z8 = {0, 0, 0, 0, 0, 0, 0, 0};
  __builtin_amdgcn_tensor_load_to_lds(g0, g1, z4, z4, z8, 0);
#else
  __builtin_amdgcn_tensor_load_to_lds(g0, g1, z4, z4, 0);
#endif
}
#endif

// ---------------- prep kernels ----------------
__global__ void gat_prep_v(const float* __restrict__ Wx, const float* __restrict__ Wa,
                           float* __restrict__ vs, float* __restrict__ vn) {
  int h = blockIdx.x, d = threadIdx.x;
  float s = 0.f, n = 0.f;
  for (int f = 0; f < FDIM; ++f) {
    float w = Wx[(h * FDIM + f) * DIM + d];
    s += w * Wa[h * 2 * FDIM + f];
    n += w * Wa[h * 2 * FDIM + FDIM + f];
  }
  vs[h * DIM + d] = s;
  vn[h * DIM + d] = n;
}

// W2t[f, h*128+d] = 0.25 * Wx[h,f,d]  (f16)
__global__ void gat_prep_w2t(const float* __restrict__ Wx, _Float16* __restrict__ w2t) {
  int idx = blockIdx.x * blockDim.x + threadIdx.x;
  if (idx >= HEADS * FDIM * DIM) return;
  int h = idx / (FDIM * DIM);
  int r = idx % (FDIM * DIM);
  int f = r / DIM;
  int d = r % DIM;
  w2t[f * KTOT + h * DIM + d] = (_Float16)(Wx[idx] * 0.25f);
}

// ---------------- fused main kernel ----------------
__global__ void __launch_bounds__(NTHR)
gat_main(const float* __restrict__ selves, const float* __restrict__ neighbors,
         const float* __restrict__ vs_g, const float* __restrict__ vn_g,
         const _Float16* __restrict__ w2t, float* __restrict__ out) {
  extern __shared__ float smem[];
  const int tid = threadIdx.x;
  const int b0 = blockIdx.x * TILE_B;

  float* nb    = smem + NB_F;
  float* selfT = smem + SELF_F;
  float* vs    = smem + VS_F;
  float* vn    = smem + VN_F;
  float* ndot  = smem + NDOT_F;
  float* scS   = smem + SCS_F;
  float* snS   = smem + SNS_F;
  float* alpha = smem + ALPHA_F;
  _Float16* yh = (_Float16*)(smem + YH_F);
  float* outT  = smem + OUT_F;
  float* rn    = smem + RN_F;

  // stage score vectors (vectorized: 256 float4s)
  if (tid < 256) {
    if (tid < 128) ((f32x4*)vs)[tid] = ((const f32x4*)vs_g)[tid];
    else           ((f32x4*)vn)[tid - 128] = ((const f32x4*)vn_g)[tid - 128];
  }

  // stage 512x128 neighbor tile + 16x128 self tile into LDS (single HBM read)
#ifdef USE_TDM
  if (tid < 32) {  // wave 0 issues TDM DMA; per-wave issue, EXEC ignored
    tdm_load_tile_f32((unsigned)(uintptr_t)nb,
                      neighbors + (size_t)b0 * KNBR * DIM, ROWS);
    tdm_load_tile_f32((unsigned)(uintptr_t)selfT,
                      selves + (size_t)b0 * DIM, TILE_B);
    __builtin_amdgcn_s_wait_tensorcnt(0);
  }
#else
  for (int i = tid; i < ROWS * DIM; i += NTHR)
    nb[(i >> 7) * ROWP + (i & 127)] = neighbors[(size_t)b0 * KNBR * DIM + i];
  for (int i = tid; i < TILE_B * DIM; i += NTHR)
    selfT[(i >> 7) * ROWP + (i & 127)] = selves[(size_t)b0 * DIM + i];
#endif
  __syncthreads();

  // ---- score dots: one neighbor row per thread, float4 LDS traffic ----
  {
    const f32x4* row4 = (const f32x4*)(nb + tid * ROWP);
    const f32x4* vn4  = (const f32x4*)vn;
    f32x4 a0 = {0.f, 0.f, 0.f, 0.f}, a1 = a0, a2 = a0, a3 = a0;
    for (int q = 0; q < DIM / 4; ++q) {
      f32x4 x = row4[q];
      a0 += x * vn4[q];
      a1 += x * vn4[32 + q];
      a2 += x * vn4[64 + q];
      a3 += x * vn4[96 + q];
    }
    ndot[tid * HEADS + 0] = a0.x + a0.y + a0.z + a0.w;
    ndot[tid * HEADS + 1] = a1.x + a1.y + a1.z + a1.w;
    ndot[tid * HEADS + 2] = a2.x + a2.y + a2.z + a2.w;
    ndot[tid * HEADS + 3] = a3.x + a3.y + a3.z + a3.w;
  }
  if (tid < TILE_B) {  // self-row dots vs both v_s and v_n
    const f32x4* row4 = (const f32x4*)(selfT + tid * ROWP);
    const f32x4* vs4  = (const f32x4*)vs;
    const f32x4* vn4  = (const f32x4*)vn;
    f32x4 s[HEADS], n[HEADS];
#pragma unroll
    for (int h = 0; h < HEADS; ++h) { s[h] = (f32x4){0.f,0.f,0.f,0.f}; n[h] = s[h]; }
    for (int q = 0; q < DIM / 4; ++q) {
      f32x4 x = row4[q];
#pragma unroll
      for (int h = 0; h < HEADS; ++h) {
        s[h] += x * vs4[h * 32 + q];
        n[h] += x * vn4[h * 32 + q];
      }
    }
#pragma unroll
    for (int h = 0; h < HEADS; ++h) {
      scS[tid * HEADS + h] = s[h].x + s[h].y + s[h].z + s[h].w;
      snS[tid * HEADS + h] = n[h].x + n[h].y + n[h].z + n[h].w;
    }
  }
  __syncthreads();

  // ---- leaky_relu + softmax over 33 entries, per (b,h) ----
  if (tid < TILE_B * HEADS) {
    int b = tid >> 2, h = tid & 3;
    float base = scS[b * HEADS + h];
    float mx = -1e30f;
    for (int k = 0; k < KP1; ++k) {
      float dv = (k < KNBR) ? ndot[(b * KNBR + k) * HEADS + h] : snS[b * HEADS + h];
      float x = base + dv;
      x = (x > 0.f) ? x : 0.2f * x;
      mx = fmaxf(mx, x);
    }
    float sum = 0.f;
    float* al = alpha + (b * HEADS + h) * KP1;
    for (int k = 0; k < KP1; ++k) {
      float dv = (k < KNBR) ? ndot[(b * KNBR + k) * HEADS + h] : snS[b * HEADS + h];
      float x = base + dv;
      x = (x > 0.f) ? x : 0.2f * x;
      float e = __expf(x - mx);
      al[k] = e;
      sum += e;
    }
    float inv = 1.f / sum;
    for (int k = 0; k < KP1; ++k) al[k] *= inv;
  }
  __syncthreads();

  // ---- aggregate in input space, ALL 4 heads at once (rows read ONCE) ----
  // thread = (b, 4-float chunk c); y[h] accumulates alpha-weighted chunk
  {
    int b = tid >> 5;            // 0..15 (one wave per b)
    int c = tid & 31;            // chunk 0..31, d0 = c*4
    f32x4 y[HEADS];
#pragma unroll
    for (int h = 0; h < HEADS; ++h) y[h] = (f32x4){0.f, 0.f, 0.f, 0.f};
    const float* al = alpha + b * HEADS * KP1;
    for (int k = 0; k < KP1; ++k) {
      const f32x4* src = (k < KNBR)
          ? (const f32x4*)(nb + (b * KNBR + k) * ROWP)
          : (const f32x4*)(selfT + b * ROWP);
      f32x4 x = src[c];
#pragma unroll
      for (int h = 0; h < HEADS; ++h) {
        float a = al[h * KP1 + k];      // LDS broadcast (same addr across wave)
        y[h] += a * x;
      }
    }
#pragma unroll
    for (int h = 0; h < HEADS; ++h) {
      f16x4 p;
      p.x = (_Float16)y[h].x; p.y = (_Float16)y[h].y;
      p.z = (_Float16)y[h].z; p.w = (_Float16)y[h].w;
      *(f16x4*)(yh + b * YSTR + h * DIM + c * 4) = p;   // 8B-aligned b64 store
    }
  }
  __syncthreads();

  // ---- final GEMM: out16x128 = Yh(16x512) @ W2t'(512x128) via WMMA f16 ----
  {
    int wv = tid >> 5;
    int ln = tid & 31;
    if (wv < 8) {                      // wave wv owns N-tile columns [wv*16, +16)
      int m  = ln & 15;                // A row (b index in tile)
      int hi = ln >> 4;                // lane-half select
      int fcol = wv * 16 + (ln & 15);  // B/D column
      v8f c = {};
#pragma unroll 4
      for (int ks = 0; ks < 16; ++ks) {
        int kb = ks * 32;
        v16h a;
#pragma unroll
        for (int i = 0; i < 8; ++i) {  // A 16x32 f16 fragment (wave32): contiguous k = 16*hi + s, same as B
          int g = i >> 2, j = i & 3;
          int koff = kb + 16 * hi + 2 * i;
          a[2 * i]     = yh[m * YSTR + koff];
          a[2 * i + 1] = yh[m * YSTR + koff + 1];
        }
        // B fragment is 16 contiguous halfs: one 32-byte load
        v16h bf = *(const v16h*)(w2t + fcol * KTOT + kb + 16 * hi);
        c = __builtin_amdgcn_wmma_f32_16x16x32_f16(false, a, false, bf,
                                                   (short)0, c, false, false);
      }
#pragma unroll
      for (int r = 0; r < 8; ++r) {    // C/D: VGPR r holds row r + 8*hi
        int bb = r + 8 * hi;
        float v = c[r];
        outT[bb * FDIM + fcol] = (v > 0.f) ? v : 0.f;  // relu
      }
    }
  }
  __syncthreads();

  // ---- L2 normalize per row, store (vectorized) ----
  if (tid < TILE_B) {
    const f32x4* o4 = (const f32x4*)(outT + tid * FDIM);
    f32x4 ssv = {0.f, 0.f, 0.f, 0.f};
    for (int q = 0; q < FDIM / 4; ++q) { f32x4 v = o4[q]; ssv += v * v; }
    float ss = ssv.x + ssv.y + ssv.z + ssv.w;
    rn[tid] = 1.f / fmaxf(sqrtf(ss), 1e-12f);
  }
  __syncthreads();
  {
    // 2048 floats = 512 float4s: exactly one per thread, coalesced b128 store
    f32x4 v = ((const f32x4*)outT)[tid];
    float r = rn[tid >> 5];
    ((f32x4*)(out + (size_t)b0 * FDIM))[tid] = v * r;
  }
}

// ---------------- host launcher ----------------
extern "C" void kernel_launch(void* const* d_in, const int* in_sizes, int n_in,
                              void* d_out, int out_size, void* d_ws, size_t ws_size,
                              hipStream_t stream) {
  const float* selves    = (const float*)d_in[0];
  const float* neighbors = (const float*)d_in[1];
  const float* Wx        = (const float*)d_in[2];
  const float* Wa        = (const float*)d_in[3];
  float* ws  = (float*)d_ws;
  float* vs  = ws;                             // 512 f32
  float* vn  = ws + KTOT;                      // 512 f32
  _Float16* w2t = (_Float16*)(ws + 2 * KTOT);  // 512x128 f16 packed weights
  float* out = (float*)d_out;

  gat_prep_v<<<HEADS, DIM, 0, stream>>>(Wx, Wa, vs, vn);
  gat_prep_w2t<<<(HEADS * FDIM * DIM + 255) / 256, 256, 0, stream>>>(Wx, w2t);
  gat_main<<<B_TOT / TILE_B, NTHR, SMEM_BYTES, stream>>>(selves, neighbors, vs, vn,
                                                         w2t, out);
}
